// LancetBlockFull_1941325218210
// MI455X (gfx1250) — matrix operand order repaired and checked by reading, not compiled, not run on hardware
//
#include <hip/hip_runtime.h>
#include <hip/hip_bf16.h>
#include <math.h>
#include <stdint.h>

typedef __bf16 bf16;
typedef __attribute__((ext_vector_type(16))) __bf16 v16bf;
typedef __attribute__((ext_vector_type(8)))  __bf16 v8bf;
typedef __attribute__((ext_vector_type(8)))  float  v8f;
typedef __attribute__((ext_vector_type(4)))  unsigned int v4u;
typedef __attribute__((ext_vector_type(8)))  int v8i;
typedef __attribute__((ext_vector_type(4)))  int v4i;

#define BM 128
#define BN 128
#define BK 32
#define LDK 40   // padded LDS K-stride (80B rows: 16B-aligned chunks, conflict-break)

// Tensor Data Mover availability (this toolchain: 6-arg builtin, per compile probe)
#if defined(__has_builtin)
#if __has_builtin(__builtin_amdgcn_tensor_load_to_lds) && \
    __has_builtin(__builtin_amdgcn_s_wait_tensorcnt)
#define HAVE_TDM 1
#endif
#endif
#ifndef HAVE_TDM
#define HAVE_TDM 0
#endif

__device__ __forceinline__ float gelu_exact(float v) {
    return 0.5f * v * (1.0f + erff(v * 0.70710678118654752f));
}

#if HAVE_TDM
// Issue a TDM load of a BM x BK tile of bf16 from a row-major [rows, K] tensor
// into LDS at lds_off, padded 4 DWORDs every 16 DWORDs -> LDK(=40)-element rows.
__device__ __forceinline__ void tdm_load_a_tile(const bf16* gsrc, unsigned lds_off,
                                                int K, int rowsTotal) {
    unsigned long long ga = (unsigned long long)(uintptr_t)gsrc;
    v4u g0;
    g0[0] = 1u;                                                 // count=1, user D#
    g0[1] = lds_off;                                            // lds_addr (bytes)
    g0[2] = (unsigned)(ga & 0xFFFFFFFFu);                       // global_addr[31:0]
    g0[3] = (unsigned)((ga >> 32) & 0x1FFFFFFu) | (2u << 30);   // ga[56:32] | type=2
    v8i g1;
    unsigned w0 = (1u << 16)    // data_size: 2 bytes
                | (1u << 20)    // pad_enable
                | (3u << 22)    // pad_interval: every 16 DWORDs (64B = one 32-elem row)
                | (3u << 25);   // pad_amount: 4 DWORDs (8 bf16) -> 40-elem LDS stride
    unsigned td0 = (unsigned)K;           // tensor_dim0 (elements per row)
    unsigned td1 = (unsigned)rowsTotal;   // tensor_dim1 (rows)
    g1[0] = (int)w0;
    g1[1] = (int)((td0 & 0xFFFFu) << 16);                 // [15:0] atomic_addr=0, td0 lo
    g1[2] = (int)((td0 >> 16) | ((td1 & 0xFFFFu) << 16)); // td0 hi | td1 lo
    g1[3] = (int)((td1 >> 16) | ((unsigned)BK << 16));    // td1 hi | tile_dim0=32
    g1[4] = (int)(unsigned)BM;                            // tile_dim1=128, tile_dim2=0
    g1[5] = (int)(unsigned)K;                             // tensor_dim0_stride[31:0]
    g1[6] = 0;                                            // stride[47:32]=0
    g1[7] = 0;
    v4i z4 = {0, 0, 0, 0};                                // 2-D tensor: groups 2/3 unused
    v8i z8 = {0, 0, 0, 0, 0, 0, 0, 0};
    __builtin_amdgcn_tensor_load_to_lds(g0, g1, z4, z4, z8, 0);
}
#endif

// ---------------- LayerNorm row statistics ----------------
__global__ void ln_stats_kernel(const float* __restrict__ x,
                                float* __restrict__ mu,
                                float* __restrict__ rstd, int D) {
    __shared__ float s0[256], s1[256];
    const int row = blockIdx.x;
    const float* p = x + (size_t)row * D;
    float s = 0.f, q = 0.f;
    for (int i = threadIdx.x; i < D; i += 256) { float v = p[i]; s += v; q += v * v; }
    s0[threadIdx.x] = s; s1[threadIdx.x] = q;
    __syncthreads();
    for (int off = 128; off > 0; off >>= 1) {
        if (threadIdx.x < off) {
            s0[threadIdx.x] += s0[threadIdx.x + off];
            s1[threadIdx.x] += s1[threadIdx.x + off];
        }
        __syncthreads();
    }
    if (threadIdx.x == 0) {
        float mean = s0[0] / (float)D;
        float var  = s1[0] / (float)D - mean * mean;
        mu[row]   = mean;
        rstd[row] = rsqrtf(var + 1e-5f);
    }
}

// ---------------- Fused bf16-WMMA GEMM ----------------
// AMODE: 0 = A from f32 x with fused LayerNorm (manual stage),
//        1 = A from bf16 buffer (TDM async-tensor stage when available)
// BTRANS: 1 = B source is W[N,K] (use W^T), 0 = B source is W[K,N]
// EPI: 0 = +bias +residual -> bf16 ; 1 = +bias, GELU -> bf16
//      2 = +bias -> bf16 ; 3 = +bias, GELU -> f32
template<int AMODE, int BTRANS, int EPI>
__global__ __launch_bounds__(256) void gemm_bf16_wmma(
    const float* __restrict__ Af32,
    const bf16*  __restrict__ Abf,
    const float* __restrict__ mu,
    const float* __restrict__ rstd,
    const float* __restrict__ lns,
    const float* __restrict__ lnb,
    const float* __restrict__ Bsrc,
    const float* __restrict__ biasBase,
    const float* __restrict__ resid,
    bf16*  __restrict__ outB,
    float* __restrict__ outF,
    int N, int K, int nKT, int Mtot,
    int chunkRows, long long bExpStride, int biasExpStride)
{
    __shared__ __align__(16) bf16 As[2][BM][LDK];
    __shared__ __align__(16) bf16 Bs[2][BN][LDK];

    const int tid  = threadIdx.x;
    const int lane = tid & 31;          // wave32
    const int wave = tid >> 5;          // 8 waves
    const int wm   = wave >> 1;         // 0..3 -> 32-row strips
    const int wn   = wave & 1;          // 0..1 -> 64-col strips
    const int l16  = lane & 15;
    const int lh   = lane >> 4;         // lane half (WMMA operand layout)

    const int mBase = blockIdx.z * chunkRows + blockIdx.y * BM;
    const int nBase = blockIdx.x * BN;
    const int e     = blockIdx.z & 7;   // expert id for chunked GEMMs
    const float* Bp   = Bsrc + (long long)e * bExpStride;
    const float* bias = biasBase + (long long)e * biasExpStride;

#if HAVE_TDM
    const bool useTdmA = (AMODE == 1);
#else
    const bool useTdmA = false;
#endif
    const bool tdmWave = (wave == 0);   // one wave drives the tensor DMA

    v8f acc[2][4];
    for (int mf = 0; mf < 2; ++mf)
        for (int nf = 0; nf < 4; ++nf)
            for (int t = 0; t < 8; ++t) acc[mf][nf][t] = 0.f;

    float aReg[16], bReg[16];

    auto stage_a = [&](int kt) {
        if (useTdmA) return;            // A staged by TDM instead
        const int k0 = kt * BK;
        #pragma unroll
        for (int i = 0; i < 16; ++i) {  // 128x32 A elements
            const int idx = i * 256 + tid;
            const int r = idx >> 5, k = idx & 31;
            const int grow = mBase + r;
            if (AMODE == 0) {
                float v = Af32[(size_t)grow * K + k0 + k];
                v = (v - mu[grow]) * rstd[grow] * lns[k0 + k] + lnb[k0 + k];
                aReg[i] = v;
            } else {
                aReg[i] = (float)Abf[(size_t)grow * K + k0 + k];
            }
        }
    };

    auto stage_a_tdm = [&](int kt, int buf) {
#if HAVE_TDM
        if (AMODE == 1 && tdmWave) {
            const bf16* src = Abf + (size_t)mBase * K + (size_t)kt * BK;
            unsigned lds_off = (unsigned)(uintptr_t)&As[buf][0][0];
            tdm_load_a_tile(src, lds_off, K, Mtot);
        }
#endif
    };

    auto stage_b = [&](int kt) {
        const int k0 = kt * BK;
        #pragma unroll
        for (int i = 0; i < 16; ++i) {  // 32x128 B elements
            const int idx = i * 256 + tid;
            if (BTRANS) {
                const int n = idx >> 5, k = idx & 31;
                bReg[i] = Bp[(size_t)(nBase + n) * K + k0 + k];
            } else {
                const int k = idx >> 7, n = idx & 127;
                bReg[i] = Bp[(size_t)(k0 + k) * N + nBase + n];
            }
        }
    };

    auto lds_store = [&](int buf) {
        if (!useTdmA) {
            #pragma unroll
            for (int i = 0; i < 16; ++i) {
                const int idx = i * 256 + tid;
                const int r = idx >> 5, k = idx & 31;
                As[buf][r][k] = (bf16)aReg[i];
            }
        }
        #pragma unroll
        for (int i = 0; i < 16; ++i) {
            const int idx = i * 256 + tid;
            if (BTRANS) {
                const int n = idx >> 5, k = idx & 31;
                Bs[buf][n][k] = (bf16)bReg[i];      // already N-major
            } else {
                const int k = idx >> 7, n = idx & 127;
                Bs[buf][n][k] = (bf16)bReg[i];      // transpose into N-major
            }
        }
    };

    auto tdm_wait = [&]() {
#if HAVE_TDM
        if (AMODE == 1 && tdmWave) __builtin_amdgcn_s_wait_tensorcnt(0);
#endif
    };

    auto compute = [&](int buf) {
        v16bf af[2], bfv[4];
        // A frag 16x32: lanes 0-15 = rows, K chunks at lh*8 and lh*8+16 (ISA layout)
        #pragma unroll
        for (int mf = 0; mf < 2; ++mf) {
            const bf16* p = &As[buf][wm * 32 + mf * 16 + l16][lh * 8];
            v8bf lo = *(const v8bf*)p;
            v8bf hi = *(const v8bf*)(p + 16);
            af[mf] = __builtin_shufflevector(lo, hi, 0, 1, 2, 3, 4, 5, 6, 7,
                                                     8, 9, 10, 11, 12, 13, 14, 15);
        }
        // B frag 32x16: lane = column, K chunks at lh*8 and lh*8+16 (N-major LDS)
        #pragma unroll
        for (int nf = 0; nf < 4; ++nf) {
            const bf16* p = &Bs[buf][wn * 64 + nf * 16 + l16][lh * 8];
            v8bf lo = *(const v8bf*)p;
            v8bf hi = *(const v8bf*)(p + 16);
            bfv[nf] = __builtin_shufflevector(lo, hi, 0, 1, 2, 3, 4, 5, 6, 7,
                                                      8, 9, 10, 11, 12, 13, 14, 15);
        }
        #pragma unroll
        for (int mf = 0; mf < 2; ++mf) {
            #pragma unroll
            for (int nf = 0; nf < 4; ++nf) {
                acc[mf][nf] = __builtin_amdgcn_wmma_f32_16x16x32_bf16(
                    false, af[mf], false, bfv[nf], (short)0, acc[mf][nf], false, false);
            }
        }
    };

    // software-pipelined, double-buffered main loop
    stage_a_tdm(0, 0);
    stage_a(0);
    stage_b(0);
    lds_store(0);
    tdm_wait();
    __syncthreads();
    for (int kt = 0; kt < nKT; ++kt) {
        if (kt + 1 < nKT) {
            stage_a_tdm(kt + 1, (kt + 1) & 1);
            stage_a(kt + 1);
            stage_b(kt + 1);
        }
        compute(kt & 1);
        if (kt + 1 < nKT) {
            lds_store((kt + 1) & 1);
            tdm_wait();
        }
        __syncthreads();
    }

    // epilogue: C/D layout — VGPR t holds row (lh*8 + t), lane&15 holds column
    const int rowW = mBase + wm * 32;
    const int colW = nBase + wn * 64;
    #pragma unroll
    for (int mf = 0; mf < 2; ++mf) {
        #pragma unroll
        for (int nf = 0; nf < 4; ++nf) {
            const int col = colW + nf * 16 + l16;
            const float bv = bias[col];
            #pragma unroll
            for (int t = 0; t < 8; ++t) {
                const int row = rowW + mf * 16 + lh * 8 + t;
                float v = acc[mf][nf][t] + bv;
                if (EPI == 0) v += resid[(size_t)row * N + col];
                if (EPI == 1 || EPI == 3) v = gelu_exact(v);
                const size_t o = (size_t)row * N + col;
                if (EPI == 3) outF[o] = v;
                else          outB[o] = (bf16)v;
            }
        }
    }
}

extern "C" void kernel_launch(void* const* d_in, const int* in_sizes, int n_in,
                              void* d_out, int out_size, void* d_ws, size_t ws_size,
                              hipStream_t stream) {
    const float* x      = (const float*)d_in[0];
    const float* ln_s   = (const float*)d_in[1];
    const float* ln_b   = (const float*)d_in[2];
    const float* attn_w = (const float*)d_in[3];
    const float* attn_b = (const float*)d_in[4];
    // d_in[5] = gate_w: dead — top-k result is discarded by the reference.
    const float* w1     = (const float*)d_in[6];
    const float* b1     = (const float*)d_in[7];
    const float* w2     = (const float*)d_in[8];
    const float* b2     = (const float*)d_in[9];
    const float* next_w = (const float*)d_in[10];
    const float* next_b = (const float*)d_in[11];
    float* out = (float*)d_out;

    const int M = 16384, D = 1024, H = 4096;

    // workspace layout (~192.2 MiB)
    char* ws = (char*)d_ws;
    float* mu    = (float*)ws;                                         // 64 KiB
    float* rstd  = (float*)(ws + (size_t)M * 4);                       // 64 KiB
    bf16*  xattn = (bf16*)(ws + (size_t)M * 8);                        // 32 MiB
    bf16*  h     = (bf16*)(ws + (size_t)M * 8 + (size_t)M * D * 2);    // 128 MiB
    bf16*  eo    = (bf16*)(ws + (size_t)M * 8 + (size_t)M * D * 2
                              + (size_t)M * H * 2);                    // 32 MiB

    dim3 blk(256);

    // 1) LayerNorm statistics per token row
    ln_stats_kernel<<<M, blk, 0, stream>>>(x, mu, rstd, D);

    // 2) x_attn = LN(x) @ attn_w^T + attn_b + x       -> bf16
    gemm_bf16_wmma<0, 1, 0><<<dim3(D / BN, M / BM, 1), blk, 0, stream>>>(
        x, nullptr, mu, rstd, ln_s, ln_b, attn_w, attn_b, x,
        xattn, nullptr, D, D, D / BK, M, 0, 0, 0);

    // 3) h = gelu(x_attn[chunk] @ w1[e] + b1[e])       -> bf16  (16 chunks of 1024 rows)
    gemm_bf16_wmma<1, 0, 1><<<dim3(H / BN, 1024 / BM, 16), blk, 0, stream>>>(
        nullptr, xattn, nullptr, nullptr, nullptr, nullptr, w1, b1, nullptr,
        h, nullptr, H, D, D / BK, M, 1024, (long long)D * H, H);

    // 4) eo = h[chunk] @ w2[e] + b2[e]                 -> bf16
    gemm_bf16_wmma<1, 0, 2><<<dim3(D / BN, 1024 / BM, 16), blk, 0, stream>>>(
        nullptr, h, nullptr, nullptr, nullptr, nullptr, w2, b2, nullptr,
        eo, nullptr, D, H, H / BK, M, 1024, (long long)H * D, D);

    // 5) out = gelu(eo @ next_w^T + next_b)            -> f32
    gemm_bf16_wmma<1, 1, 3><<<dim3(D / BN, M / BM, 1), blk, 0, stream>>>(
        nullptr, eo, nullptr, nullptr, nullptr, nullptr, next_w, next_b, nullptr,
        nullptr, out, D, D, D / BK, M, 0, 0, 0);
}
